// VerticalModel_GIN_GT_71665824301898
// MI455X (gfx1250) — compile-verified, matrix operand order checked by reading, not run on hardware
//
#include <hip/hip_runtime.h>
#include <hip/hip_bf16.h>
#include <math.h>

typedef __attribute__((ext_vector_type(16))) __bf16 v16bf;
typedef __attribute__((ext_vector_type(8)))  __bf16 v8bf;
typedef __attribute__((ext_vector_type(8)))  float  v8f;
typedef __attribute__((ext_vector_type(4)))  float  v4f;

#define CDIV(a, b) (((a) + (b) - 1) / (b))
#define LDW 136  // LDS row stride (bf16 elems): 272B keeps 16B alignment for k0 multiples of 8

// C[M x (NT*16)] = act(A[M x Kin] @ W[Kin x (NT*16)] + bias)
// bf16 WMMA with f32 accumulate. W transposed+converted into LDS, K zero-padded to mult of 32.
// Guards on A chunk loads are wave-uniform for all shapes used here (Kin mult of 16, k0 mult
// of 32), so they compile to scalar branches, not EXEC masking.
template <int NT>
__global__ __launch_bounds__(256) void gemm_bias_act(
    const float* __restrict__ A, const float* __restrict__ W,
    const float* __restrict__ bias, float* __restrict__ C,
    int M, int Kin, int act)
{
  const int Kout = NT * 16;
  const int Kpad = (Kin + 31) & ~31;
  __shared__ __bf16 sWT[128 * LDW];

  for (int idx = threadIdx.x; idx < Kout * Kpad; idx += blockDim.x) {
    int n = idx / Kpad, kk = idx - n * Kpad;
    float w = (kk < Kin) ? W[(size_t)kk * Kout + n] : 0.0f;
    sWT[n * LDW + kk] = (__bf16)w;
  }
  __syncthreads();

  const int wave  = threadIdx.x >> 5;
  const int lane  = threadIdx.x & 31;
  const int l15   = lane & 15;
  const int khalf = lane >> 4;
  const int m0    = blockIdx.x * 128 + wave * 16;

  int rowA = m0 + l15;
  if (rowA >= M) rowA = M - 1;
  const float* Arow = A + (size_t)rowA * Kin;

  v8f acc[NT];
#pragma unroll
  for (int t = 0; t < NT; ++t)
#pragma unroll
    for (int i = 0; i < 8; ++i) acc[t][i] = 0.0f;

  for (int k0 = 0; k0 < Kpad; k0 += 32) {
    const int ka = k0 + khalf * 8;
    const int kb = k0 + 16 + khalf * 8;
    const bool okA = (k0 + 16 <= Kin);  // uniform across the wave
    const bool okB = (k0 + 32 <= Kin);  // uniform across the wave

    v4f a0 = {}, a1 = {}, a2 = {}, a3 = {};
    if (okA) { a0 = *(const v4f*)(Arow + ka); a1 = *(const v4f*)(Arow + ka + 4); }
    if (okB) { a2 = *(const v4f*)(Arow + kb); a3 = *(const v4f*)(Arow + kb + 4); }

    v16bf afrag;
#pragma unroll
    for (int i = 0; i < 4; ++i) {
      afrag[i]      = (__bf16)a0[i];
      afrag[4 + i]  = (__bf16)a1[i];
      afrag[8 + i]  = (__bf16)a2[i];
      afrag[12 + i] = (__bf16)a3[i];
    }

    // Preload all B fragments (lets the compiler clause the ds_loads and then
    // issue the NT WMMAs back-to-back).
    v16bf bfrag[NT];
#pragma unroll
    for (int t = 0; t < NT; ++t) {
      const __bf16* bp = &sWT[(t * 16 + l15) * LDW];
      v8bf b0 = *(const v8bf*)(bp + ka);
      v8bf b1 = *(const v8bf*)(bp + kb);
#pragma unroll
      for (int i = 0; i < 8; ++i) { bfrag[t][i] = b0[i]; bfrag[t][8 + i] = b1[i]; }
    }
#pragma unroll
    for (int t = 0; t < NT; ++t) {
      acc[t] = __builtin_amdgcn_wmma_f32_16x16x32_bf16(
          false, afrag, false, bfrag[t], (short)0, acc[t], false, false);
    }
  }

#pragma unroll
  for (int t = 0; t < NT; ++t) {
    const int col = t * 16 + l15;
    const float b = bias[col];
#pragma unroll
    for (int r = 0; r < 8; ++r) {
      int row = m0 + khalf * 8 + r;
      if (row < M) {
        float vv = acc[t][r] + b;
        if (act) vv = fmaxf(vv, 0.0f);
        C[(size_t)row * Kout + col] = vv;
      }
    }
  }
}

// agg[dst] += h[src], 4 features per thread
__global__ void scatter_add4(const float* __restrict__ h, const int* __restrict__ src,
                             const int* __restrict__ dst, float* __restrict__ agg,
                             int E, int d)
{
  const int c = d >> 2;
  size_t i = (size_t)blockIdx.x * blockDim.x + threadIdx.x;
  if (i >= (size_t)E * c) return;
  int eid = (int)(i / c);
  int c4  = (int)(i - (size_t)eid * c) * 4;
  int s = src[eid], t = dst[eid];
  __builtin_prefetch(h + (size_t)s * d + c4, 0, 0);
  float4 vs = *(const float4*)(h + (size_t)s * d + c4);
  float* p = agg + (size_t)t * d + c4;
  atomicAdd(p + 0, vs.x); atomicAdd(p + 1, vs.y);
  atomicAdd(p + 2, vs.z); atomicAdd(p + 3, vs.w);
}

__device__ __forceinline__ unsigned enc_f32(float f) {
  unsigned u = __float_as_uint(f);
  return (u & 0x80000000u) ? ~u : (u | 0x80000000u);
}
__device__ __forceinline__ float dec_f32(unsigned u) {
  return (u & 0x80000000u) ? __uint_as_float(u ^ 0x80000000u) : __uint_as_float(~u);
}

// alpha[e,h] = dot(q[dst], k[src]+e[e]) / 8 ; segment-max via ordered-uint atomicMax
__global__ void edge_alpha(const float* __restrict__ q, const float* __restrict__ k,
                           const float* __restrict__ e, const int* __restrict__ src,
                           const int* __restrict__ dst, float* __restrict__ alpha,
                           unsigned* __restrict__ mEnc, int E)
{
  size_t i = (size_t)blockIdx.x * blockDim.x + threadIdx.x;
  if (i >= (size_t)E * 2) return;
  int eid = (int)(i >> 1), hh = (int)(i & 1);
  int sN = src[eid], dN = dst[eid];
  const float* qd = q + (size_t)dN * 128 + hh * 64;
  const float* ks = k + (size_t)sN * 128 + hh * 64;
  const float* ee = e + (size_t)eid * 128 + hh * 64;
  float s = 0.0f;
#pragma unroll 4
  for (int j = 0; j < 64; j += 4) {
    float4 a = *(const float4*)(qd + j);
    float4 b = *(const float4*)(ks + j);
    float4 c = *(const float4*)(ee + j);
    s += a.x * (b.x + c.x) + a.y * (b.y + c.y) + a.z * (b.z + c.z) + a.w * (b.w + c.w);
  }
  s *= 0.125f;  // 1/sqrt(64)
  alpha[i] = s;
  atomicMax(&mEnc[(size_t)dN * 2 + hh], enc_f32(s));
}

__global__ void edge_expsum(const float* __restrict__ alpha, const unsigned* __restrict__ mEnc,
                            const int* __restrict__ dst, float* __restrict__ ex,
                            float* __restrict__ denom, int E)
{
  size_t i = (size_t)blockIdx.x * blockDim.x + threadIdx.x;
  if (i >= (size_t)E * 2) return;
  int eid = (int)(i >> 1), hh = (int)(i & 1);
  int dN = dst[eid];
  float m = dec_f32(mEnc[(size_t)dN * 2 + hh]);
  float v = expf(alpha[i] - m);
  ex[i] = v;
  atomicAdd(&denom[(size_t)dN * 2 + hh], v);
}

// out[dst] += (v[src] + e) * (ex/denom[dst]); 4 features per thread over 128-wide rows
__global__ void edge_wsum(const float* __restrict__ v, const float* __restrict__ e,
                          const float* __restrict__ ex, const float* __restrict__ denom,
                          const int* __restrict__ src, const int* __restrict__ dst,
                          float* __restrict__ out, int E)
{
  size_t i = (size_t)blockIdx.x * blockDim.x + threadIdx.x;
  if (i >= (size_t)E * 32) return;
  int eid = (int)(i >> 5);
  int c4  = (int)(i & 31) * 4;
  int hh  = c4 >> 6;
  int sN = src[eid], dN = dst[eid];
  float a = ex[(size_t)eid * 2 + hh] / denom[(size_t)dN * 2 + hh];
  __builtin_prefetch(v + (size_t)sN * 128 + c4, 0, 0);
  float4 vv = *(const float4*)(v + (size_t)sN * 128 + c4);
  float4 ee = *(const float4*)(e + (size_t)eid * 128 + c4);
  float* p = out + (size_t)dN * 128 + c4;
  atomicAdd(p + 0, (vv.x + ee.x) * a);
  atomicAdd(p + 1, (vv.y + ee.y) * a);
  atomicAdd(p + 2, (vv.z + ee.z) * a);
  atomicAdd(p + 3, (vv.w + ee.w) * a);
}

// beta = sigmoid([out, x_r, out - x_r] @ Wbeta); h = beta*x_r + (1-beta)*out
__global__ void beta_blend(const float* __restrict__ ao, const float* __restrict__ xr,
                           const float* __restrict__ Wb, float* __restrict__ hout, int N)
{
  int n = blockIdx.x * blockDim.x + threadIdx.x;
  if (n >= N) return;
  const float* o = ao + (size_t)n * 128;
  const float* r = xr + (size_t)n * 128;
  float s = 0.0f;
#pragma unroll 4
  for (int j = 0; j < 128; j += 4) {
    float4 a  = *(const float4*)(o + j);
    float4 b  = *(const float4*)(r + j);
    float4 w0 = *(const float4*)(Wb + j);
    float4 w1 = *(const float4*)(Wb + 128 + j);
    float4 w2 = *(const float4*)(Wb + 256 + j);
    s += a.x * w0.x + b.x * w1.x + (a.x - b.x) * w2.x;
    s += a.y * w0.y + b.y * w1.y + (a.y - b.y) * w2.y;
    s += a.z * w0.z + b.z * w1.z + (a.z - b.z) * w2.z;
    s += a.w * w0.w + b.w * w1.w + (a.w - b.w) * w2.w;
  }
  float beta = 1.0f / (1.0f + expf(-s));
  float* hp = hout + (size_t)n * 128;
#pragma unroll 4
  for (int j = 0; j < 128; ++j) hp[j] = beta * r[j] + (1.0f - beta) * o[j];
}

__global__ void pool_kernel(const float* __restrict__ h, const int* __restrict__ batch,
                            float* __restrict__ pooled, float* __restrict__ cnt, int N)
{
  size_t i = (size_t)blockIdx.x * blockDim.x + threadIdx.x;
  if (i >= (size_t)N * 32) return;
  int n  = (int)(i >> 5);
  int c4 = (int)(i & 31) * 4;
  int g = batch[n];
  float4 v = *(const float4*)(h + (size_t)n * 128 + c4);
  float* p = pooled + (size_t)g * 128 + c4;
  atomicAdd(p + 0, v.x); atomicAdd(p + 1, v.y);
  atomicAdd(p + 2, v.z); atomicAdd(p + 3, v.w);
  if ((i & 31) == 0) atomicAdd(&cnt[g], 1.0f);
}

__global__ void readout_kernel(const float* __restrict__ pooled, const float* __restrict__ cnt,
                               const float* __restrict__ w1, const float* __restrict__ b1,
                               const float* __restrict__ w2, const float* __restrict__ b2,
                               float* __restrict__ out)
{
  __shared__ float sh[64];
  int g = blockIdx.x, t = threadIdx.x;
  float c = fmaxf(cnt[g], 1.0f);
  float inv = 1.0f / c;
  float s = 0.0f;
  for (int j = 0; j < 128; ++j) s += pooled[(size_t)g * 128 + j] * inv * w1[j * 64 + t];
  s = fmaxf(s + b1[t], 0.0f);
  sh[t] = s * w2[t];
  __syncthreads();
  if (t == 0) {
    float r = 0.0f;
    for (int j = 0; j < 64; ++j) r += sh[j];
    out[g] = r + b2[0];
  }
}

extern "C" void kernel_launch(void* const* d_in, const int* in_sizes, int n_in,
                              void* d_out, int out_size, void* d_ws, size_t ws_size,
                              hipStream_t stream)
{
  const float* x         = (const float*)d_in[0];
  const float* edge_attr = (const float*)d_in[1];
  const int*   edge_idx  = (const int*)d_in[2];
  const int*   batch     = (const int*)d_in[3];
  const int F = 32;
  const int N = in_sizes[0] / F;
  const int E = in_sizes[2] / 2;
  const int G = 512;
  const int* srcI = edge_idx;
  const int* dstI = edge_idx + E;

  float* base = (float*)d_ws;
  size_t off = 0;
  auto alloc = [&](size_t n) { float* p = base + off; off += (n + 63) & ~(size_t)63; return p; };

  float* hA   = alloc((size_t)N * 128);
  float* hB   = alloc((size_t)N * 128);
  float* tmp  = alloc((size_t)N * 128);
  float* qb   = alloc((size_t)N * 128);
  float* kb   = alloc((size_t)N * 128);
  float* vb   = alloc((size_t)N * 128);
  float* sk   = alloc((size_t)N * 128);
  float* eb   = alloc((size_t)E * 128);
  float* al   = alloc((size_t)E * 2);
  float* exb  = alloc((size_t)E * 2);
  unsigned* mE = (unsigned*)alloc((size_t)N * 2);
  float* dn   = alloc((size_t)N * 2);
  float* ao   = alloc((size_t)N * 128);
  float* pooled = alloc((size_t)G * 128);
  float* cntb   = alloc((size_t)G);

  // ---- GIN layers ----
  hipMemcpyAsync(hA, x, (size_t)N * F * sizeof(float), hipMemcpyDeviceToDevice, stream);
  int d = F;
  for (int l = 0; l < 3; ++l) {
    const float* w1 = (const float*)d_in[4 + l * 4 + 0];
    const float* b1 = (const float*)d_in[4 + l * 4 + 1];
    const float* w2 = (const float*)d_in[4 + l * 4 + 2];
    const float* b2 = (const float*)d_in[4 + l * 4 + 3];
    // agg = h + sum_{src->dst} h[src]
    hipMemcpyAsync(hB, hA, (size_t)N * d * sizeof(float), hipMemcpyDeviceToDevice, stream);
    size_t nthr = (size_t)E * (d >> 2);
    scatter_add4<<<dim3((unsigned)CDIV(nthr, 256)), dim3(256), 0, stream>>>(hA, srcI, dstI, hB, E, d);
    gemm_bias_act<4><<<dim3(CDIV(N, 128)), dim3(256), 0, stream>>>(hB, w1, b1, tmp, N, d, 1);
    gemm_bias_act<4><<<dim3(CDIV(N, 128)), dim3(256), 0, stream>>>(tmp, w2, b2, hA, N, 64, 0);
    d = 64;
  }

  // ---- TransformerConv layers ----
  const float* hcur = hA;
  for (int l = 0; l < 2; ++l) {
    // pytree (sorted-key) order: Wbeta, We, Wk, Wq, Wskip, Wv, be, bk, bq, bskip, bv
    int bix = 16 + l * 11;
    const float* Wbeta = (const float*)d_in[bix + 0];
    const float* We    = (const float*)d_in[bix + 1];
    const float* Wk    = (const float*)d_in[bix + 2];
    const float* Wq    = (const float*)d_in[bix + 3];
    const float* Wskip = (const float*)d_in[bix + 4];
    const float* Wv    = (const float*)d_in[bix + 5];
    const float* be    = (const float*)d_in[bix + 6];
    const float* bk    = (const float*)d_in[bix + 7];
    const float* bq    = (const float*)d_in[bix + 8];
    const float* bskip = (const float*)d_in[bix + 9];
    const float* bv    = (const float*)d_in[bix + 10];

    gemm_bias_act<8><<<dim3(CDIV(N, 128)), dim3(256), 0, stream>>>(hcur, Wq, bq, qb, N, d, 0);
    gemm_bias_act<8><<<dim3(CDIV(N, 128)), dim3(256), 0, stream>>>(hcur, Wk, bk, kb, N, d, 0);
    gemm_bias_act<8><<<dim3(CDIV(N, 128)), dim3(256), 0, stream>>>(hcur, Wv, bv, vb, N, d, 0);
    gemm_bias_act<8><<<dim3(CDIV(N, 128)), dim3(256), 0, stream>>>(hcur, Wskip, bskip, sk, N, d, 0);
    gemm_bias_act<8><<<dim3(CDIV(E, 128)), dim3(256), 0, stream>>>(edge_attr, We, be, eb, E, 16, 0);

    hipMemsetAsync(mE, 0, (size_t)N * 2 * sizeof(unsigned), stream);
    edge_alpha<<<dim3((unsigned)CDIV((size_t)E * 2, 256)), dim3(256), 0, stream>>>(
        qb, kb, eb, srcI, dstI, al, mE, E);
    hipMemsetAsync(dn, 0, (size_t)N * 2 * sizeof(float), stream);
    hipMemsetAsync(ao, 0, (size_t)N * 128 * sizeof(float), stream);
    edge_expsum<<<dim3((unsigned)CDIV((size_t)E * 2, 256)), dim3(256), 0, stream>>>(
        al, mE, dstI, exb, dn, E);
    edge_wsum<<<dim3((unsigned)CDIV((size_t)E * 32, 256)), dim3(256), 0, stream>>>(
        vb, eb, exb, dn, srcI, dstI, ao, E);
    beta_blend<<<dim3(CDIV(N, 256)), dim3(256), 0, stream>>>(ao, sk, Wbeta, hB, N);
    hcur = hB;
    d = 128;
  }

  // ---- pool + readout ----
  const float* rw1 = (const float*)d_in[38];
  const float* rb1 = (const float*)d_in[39];
  const float* rw2 = (const float*)d_in[40];
  const float* rb2 = (const float*)d_in[41];
  hipMemsetAsync(pooled, 0, (size_t)G * 128 * sizeof(float), stream);
  hipMemsetAsync(cntb, 0, (size_t)G * sizeof(float), stream);
  pool_kernel<<<dim3((unsigned)CDIV((size_t)N * 32, 256)), dim3(256), 0, stream>>>(
      hcur, batch, pooled, cntb, N);
  readout_kernel<<<dim3(G), dim3(64), 0, stream>>>(pooled, cntb, rw1, rb1, rw2, rb2,
                                                   (float*)d_out);
}
